// AutoGroupVectorQuantize_45200235823582
// MI455X (gfx1250) — compile-verified, matrix operand order checked by reading, not run on hardware
//
#include <hip/hip_runtime.h>
#include <hip/hip_bf16.h>

// ---------------- problem constants ----------------
#define BB 8
#define DD 1024
#define TT 4096
#define CD 8          // codebook dim
#define CS 4096       // codebook size
#define NFRAMES (BB * TT)           // 32768
#define INV_MEAN 3.0517578125e-05f  // 1 / (CD * TT) = 1/32768

typedef __attribute__((ext_vector_type(16))) _Float16 v16h;
typedef __attribute__((ext_vector_type(8)))  float    v8f;
typedef __attribute__((ext_vector_type(2)))  float    v2f;
typedef __attribute__((ext_vector_type(4)))  int      v4i;
typedef __attribute__((ext_vector_type(4)))  float    v4f;

union U16h { int i[8]; v16h h; };
union U8h  { _Float16 h[8]; v4i q; };

// ---------------------------------------------------------------------------
// TDM: 1-D tensor DMA global -> LDS. This toolchain has the 6-arg builtin
// (uint32x4 g0, int32x8 g1, int32x4 g2, int32x4 g3, int32x8 extra, i32 cpol).
// Descriptor fields per CDNA5 ISA.
// ---------------------------------------------------------------------------
#if defined(__has_builtin)
#if __has_builtin(__builtin_amdgcn_tensor_load_to_lds)
#define HAVE_TDM 1
#endif
#endif

#if defined(HAVE_TDM)
typedef unsigned int u32x4 __attribute__((ext_vector_type(4)));
typedef int          i32x4 __attribute__((ext_vector_type(4)));
typedef int          i32x8 __attribute__((ext_vector_type(8)));

// Copy n8 * 8 bytes from gsrc to LDS (one TENSOR_LOAD_TO_LDS, wave-level op).
__device__ inline void tdm_load_1d(const void* gsrc, const void* lds_dst, unsigned n8) {
    unsigned long long ga = (unsigned long long)gsrc;
    unsigned lds_off = (unsigned)(unsigned long long)lds_dst; // low 32 bits = LDS offset
    // group0: count=1 | lds_addr | global_addr[56:0] | type=2 ("image")
    u32x4 g0 = { 1u,
                 lds_off,
                 (unsigned)(ga & 0xffffffffu),
                 (unsigned)((ga >> 32) & 0x01ffffffu) | (2u << 30) };
    // group1: data_size=3 (8B units), tensor_dim0=n8, tensor_dim1=1,
    //         tile_dim0=n8 (1-D tile), dim0_stride=n8
    i32x8 g1 = { (int)(3u << 16),                       // [17:16] data_size=3
                 (int)((n8 & 0xffffu) << 16),           // [63:48]  tensor_dim0 lo
                 (int)((n8 >> 16) | (1u << 16)),        // [95:64]  dim0 hi | tensor_dim1=1
                 (int)(n8 << 16),                       // [127:112] tile_dim0
                 0,                                     // tile_dim1/2 = 0 (unused)
                 (int)n8,                               // [207:160] tensor_dim0_stride lo
                 0, 0 };
    i32x4 gz = { 0, 0, 0, 0 };                          // groups 2/3 unused (<=2D)
    i32x8 z8 = { 0, 0, 0, 0, 0, 0, 0, 0 };              // extra group (6-arg form)
    __builtin_amdgcn_tensor_load_to_lds(g0, g1, gz, gz, z8, 0);
}
#endif

__device__ inline void lds_fill_manual(void* dst, const void* src, int n16,
                                       int tid, int nthreads) {
    const v4i* s = (const v4i*)src;
    v4i*       d = (v4i*)dst;
    for (int j = tid; j < n16; j += nthreads) d[j] = s[j];
}

// ---------------------------------------------------------------------------
// K0a: normalize both codebooks to f16 rows [CS][8]; zero the loss slots.
// ---------------------------------------------------------------------------
__global__ __launch_bounds__(256) void cbnorm_kernel(
    const float* __restrict__ cb_a, const float* __restrict__ cb_b,
    _Float16* __restrict__ cbn_a, _Float16* __restrict__ cbn_b,
    float* __restrict__ loss_slots)
{
    if (blockIdx.x == 0 && threadIdx.x < 16) loss_slots[threadIdx.x] = 0.0f;

    int g = blockIdx.x * 256 + threadIdx.x;      // 0 .. 2*CS-1
    const float* src = (g < CS) ? cb_a : cb_b;
    _Float16*    dst = (g < CS) ? cbn_a : cbn_b;
    int c = g & (CS - 1);

    const v4f* p = (const v4f*)(src + (size_t)c * CD);
    v4f lo = p[0], hi = p[1];
    float s = lo.x*lo.x + lo.y*lo.y + lo.z*lo.z + lo.w*lo.w
            + hi.x*hi.x + hi.y*hi.y + hi.z*hi.z + hi.w*hi.w;
    float r = 1.0f / fmaxf(sqrtf(s), 1e-12f);

    U8h o;
    o.h[0] = (_Float16)(lo.x * r); o.h[1] = (_Float16)(lo.y * r);
    o.h[2] = (_Float16)(lo.z * r); o.h[3] = (_Float16)(lo.w * r);
    o.h[4] = (_Float16)(hi.x * r); o.h[5] = (_Float16)(hi.y * r);
    o.h[6] = (_Float16)(hi.z * r); o.h[7] = (_Float16)(hi.w * r);
    *(v4i*)(dst + (size_t)c * CD) = o.q;
}

// ---------------------------------------------------------------------------
// K0b: pre-swizzle the combined input weights [16ch][1024] into the f32 WMMA
// A-matrix per-K-step layout:  wpack[s][lane][j] = W[lane%16][4s + j + (lane<16?0:2)]
// (A 16x4 f32: VGPR0 = K0 | K2 split across lane halves, VGPR1 = K1 | K3.)
// ---------------------------------------------------------------------------
__global__ __launch_bounds__(256) void packw_kernel(
    const float* __restrict__ w_in_a, const float* __restrict__ w_in_b,
    float* __restrict__ wpack)                        // [256][32][2] f32 = 64 KB
{
    int idx = blockIdx.x * 256 + threadIdx.x;         // 0 .. 16383
    int s = idx >> 6;
    int r = idx & 63;
    int l = r >> 1;
    int j = r & 1;
    int m = l & 15;
    int k = 4 * s + j + ((l < 16) ? 0 : 2);
    wpack[idx] = (m < 8) ? w_in_a[m * DD + k] : w_in_b[(m - 8) * DD + k];
}

// ---------------------------------------------------------------------------
// K1: input projection with full-precision f32 WMMA (V_WMMA_F32_16X16X4_F32).
// Block = 8 waves; each wave owns one 16-frame tile and accumulates the
// 16(ch) x 16(frame) latent over K=1024 in 256 accumulating WMMAs.
// C layout: lane = frame column; lanes 0..15 hold ch 0..7 (enc_a vector),
// lanes 16..31 hold ch 8..15 (enc_b vector) -> per-lane L2-normalize + f16 pack.
// ---------------------------------------------------------------------------
__global__ __launch_bounds__(256) void project_wmma_kernel(
    const float* __restrict__ z,
    const float* __restrict__ wpack,     // [256][32][2] from K0b
    const float* __restrict__ b_in_a, const float* __restrict__ b_in_b,
    float* __restrict__ latent,          // [B][16][T]
    _Float16* __restrict__ encn)         // [NFRAMES][16]
{
    __shared__ __align__(16) float s_wpack[256 * 64];     // 64 KB

    const int tid  = threadIdx.x;
    const int lane = tid & 31;
    const int wave = tid >> 5;

#if defined(HAVE_TDM)
    if (wave == 0) {
        tdm_load_1d(wpack, s_wpack, 256 * 64 * 4 / 8);
        __builtin_amdgcn_s_wait_tensorcnt(0);
    }
#else
    lds_fill_manual(s_wpack, wpack, 256 * 64 / 4, tid, 256);
#endif
    __syncthreads();

    const int g   = blockIdx.x * 8 + wave;    // tile id 0..2047
    const int b   = g >> 8;
    const int t0  = (g & 255) << 4;
    const int col = lane & 15;
    const int koff = (lane < 16) ? 0 : 2;

    const float* pz = z + ((size_t)b * DD + koff) * TT + t0 + col;
    const v2f*   wl = (const v2f*)s_wpack;

    v8f acc = {0.f, 0.f, 0.f, 0.f, 0.f, 0.f, 0.f, 0.f};
    for (int s = 0; s < DD / 4; ++s) {
        v2f A = wl[s * 32 + lane];                 // ds_load_b64
        v2f Bv; Bv.x = pz[0]; Bv.y = pz[TT];       // two rows of z, coalesced over t
        pz += (size_t)4 * TT;
        acc = __builtin_amdgcn_wmma_f32_16x16x4_f32(
            false, A, false, Bv, (short)0, acc, false, false);
    }

    // add bias (per-lane channel group), write latent, normalize, pack f16
    float v[8];
#pragma unroll
    for (int r = 0; r < 8; ++r) {
        float bias = (lane < 16) ? b_in_a[r] : b_in_b[r];
        v[r] = acc[r] + bias;
    }

    float* lp = latent + ((size_t)b * 16 + (lane < 16 ? 0 : 8)) * TT + t0 + col;
#pragma unroll
    for (int r = 0; r < 8; ++r) lp[(size_t)r * TT] = v[r];

    float n = 0.f;
#pragma unroll
    for (int r = 0; r < 8; ++r) n += v[r] * v[r];
    float rinv = 1.0f / fmaxf(sqrtf(n), 1e-12f);

    U8h e;
#pragma unroll
    for (int r = 0; r < 8; ++r) e.h[r] = (_Float16)(v[r] * rinv);
    _Float16* ep = encn + (size_t)(b * TT + t0 + col) * 16 + (lane < 16 ? 0 : 8);
    *(v4i*)ep = e.q;
}

// ---------------------------------------------------------------------------
// K2: WMMA nearest-code search, both codebooks interleaved as two independent
// WMMA->argmax streams (hides the WMMA->VALU hazard window). Both normalized
// codebooks (128 KB) staged in LDS via TDM. argmin(dist) == argmax(cos sim).
// ---------------------------------------------------------------------------
__global__ __launch_bounds__(256) void vq_argmax_kernel(
    const _Float16* __restrict__ encn,   // [NFRAMES][16]
    const _Float16* __restrict__ cbn_a,  // [CS][8]
    const _Float16* __restrict__ cbn_b,  // [CS][8]
    int* __restrict__ idx_a, int* __restrict__ idx_b)
{
    __shared__ __align__(16) _Float16 s_cba[CS * CD];  // 64 KB
    __shared__ __align__(16) _Float16 s_cbb[CS * CD];  // 64 KB
    __shared__ __align__(16) _Float16 s_zero[8];       // zero row for lanes >= 16

    const int tid  = threadIdx.x;
    const int lane = tid & 31;
    const int wave = tid >> 5;
    const int frame_base = blockIdx.x * 128 + wave * 16;

    if (tid < 8) s_zero[tid] = (_Float16)0.0f;

#if defined(HAVE_TDM)
    if (wave == 0) {
        tdm_load_1d(cbn_a, s_cba, CS * CD * 2 / 8);
        __builtin_amdgcn_s_wait_tensorcnt(0);
    } else if (wave == 1) {
        tdm_load_1d(cbn_b, s_cbb, CS * CD * 2 / 8);
        __builtin_amdgcn_s_wait_tensorcnt(0);
    }
#else
    lds_fill_manual(s_cba, cbn_a, CS * CD / 8, tid, 256);
    lds_fill_manual(s_cbb, cbn_b, CS * CD / 8, tid, 256);
#endif

    // ---- A matrices (enc_a, enc_b) for this wave's 16 frames -------------
    // A f16 16x32: lane m<16 holds K=0..7 (low 4 dword pairs); rest zero.
    U16h Aa, Ab;
#pragma unroll
    for (int k = 0; k < 8; ++k) { Aa.i[k] = 0; Ab.i[k] = 0; }
    if (lane < 16) {
        const v4i* pa = (const v4i*)(encn + (size_t)(frame_base + lane) * 16);
        v4i ra = pa[0];
        v4i rb = pa[1];
        Aa.i[0] = ra.x; Aa.i[1] = ra.y; Aa.i[2] = ra.z; Aa.i[3] = ra.w;
        Ab.i[0] = rb.x; Ab.i[1] = rb.y; Ab.i[2] = rb.z; Ab.i[3] = rb.w;
    }
    __syncthreads();

    // per-lane B pointers: lanes 0..15 walk code rows, lanes 16..31 pinned to zeros
    const _Float16 *bpa, *bpb;
    int bstride;
    if (lane < 16) { bpa = s_cba + (size_t)lane * CD;
                     bpb = s_cbb + (size_t)lane * CD; bstride = 16 * CD; }
    else           { bpa = s_zero; bpb = s_zero;      bstride = 0; }

    float bestA[8], bestB[8];
    int   idxA[8],  idxB[8];
#pragma unroll
    for (int r = 0; r < 8; ++r) {
        bestA[r] = -3.0e38f; bestB[r] = -3.0e38f; idxA[r] = 0; idxB[r] = 0;
    }

    for (int chunk = 0; chunk < CS / 16; ++chunk) {
        v4i ra = *(const v4i*)bpa; bpa += bstride;     // ds_load_b128
        v4i rb = *(const v4i*)bpb; bpb += bstride;

        U16h BA, BBm;
        BA.i[0]  = ra.x; BA.i[1]  = ra.y; BA.i[2]  = ra.z; BA.i[3]  = ra.w;
        BA.i[4]  = 0;    BA.i[5]  = 0;    BA.i[6]  = 0;    BA.i[7]  = 0;
        BBm.i[0] = rb.x; BBm.i[1] = rb.y; BBm.i[2] = rb.z; BBm.i[3] = rb.w;
        BBm.i[4] = 0;    BBm.i[5] = 0;    BBm.i[6] = 0;    BBm.i[7] = 0;

        v8f zero8 = {0.f, 0.f, 0.f, 0.f, 0.f, 0.f, 0.f, 0.f};
        v8f sA = __builtin_amdgcn_wmma_f32_16x16x32_f16(
            false, Aa.h, false, BA.h, (short)0, zero8, false, false);
        v8f sB = __builtin_amdgcn_wmma_f32_16x16x32_f16(
            false, Ab.h, false, BBm.h, (short)0, zero8, false, false);

        const int code0 = (chunk << 4) + (lane & 15);
#pragma unroll
        for (int r = 0; r < 8; ++r) {
            bool gA = sA[r] > bestA[r];
            bestA[r] = gA ? sA[r] : bestA[r];
            idxA[r]  = gA ? code0 : idxA[r];
            bool gB = sB[r] > bestB[r];
            bestB[r] = gB ? sB[r] : bestB[r];
            idxB[r]  = gB ? code0 : idxB[r];
        }
    }

    // cross-lane argmax within each 16-lane half; ties -> lower code index
#pragma unroll
    for (int s = 1; s < 16; s <<= 1) {
#pragma unroll
        for (int r = 0; r < 8; ++r) {
            float om = __shfl_xor(bestA[r], s, 32);
            int   oi = __shfl_xor(idxA[r],  s, 32);
            bool tk = (om > bestA[r]) || ((om == bestA[r]) && (oi < idxA[r]));
            bestA[r] = tk ? om : bestA[r];
            idxA[r]  = tk ? oi : idxA[r];
            float pm = __shfl_xor(bestB[r], s, 32);
            int   pi = __shfl_xor(idxB[r],  s, 32);
            bool tb = (pm > bestB[r]) || ((pm == bestB[r]) && (pi < idxB[r]));
            bestB[r] = tb ? pm : bestB[r];
            idxB[r]  = tb ? pi : idxB[r];
        }
    }
    if ((lane & 15) == 0) {
        int mrow = (lane >> 4) * 8;          // lane 0 -> M 0..7, lane 16 -> M 8..15
#pragma unroll
        for (int r = 0; r < 8; ++r) {
            idx_a[frame_base + mrow + r] = idxA[r];
            idx_b[frame_base + mrow + r] = idxB[r];
        }
    }
}

// ---------------------------------------------------------------------------
// K3: gather codebook rows (unnormalized), losses, combined indices,
// and the output projection  z_q = [w_out_a @ zq_a ; w_out_b @ zq_b].
// ---------------------------------------------------------------------------
__global__ __launch_bounds__(256) void output_kernel(
    const float* __restrict__ cb_a, const float* __restrict__ cb_b,
    const float* __restrict__ w_out_a, const float* __restrict__ b_out_a,
    const float* __restrict__ w_out_b, const float* __restrict__ b_out_b,
    const int* __restrict__ idx_a, const int* __restrict__ idx_b,
    const float* __restrict__ latent,
    float* __restrict__ zq_out,
    float* __restrict__ commit_loss, float* __restrict__ cbk_loss,
    float* __restrict__ indices_out)
{
    __shared__ float s_zq[16][64];
    const int tid = threadIdx.x;
    const int b   = blockIdx.y;
    const int t0  = blockIdx.x * 64;

    float lsum = 0.0f;
    if (tid < 64) {
        const int t     = t0 + tid;
        const int frame = b * TT + t;
        const int ia = idx_a[frame];
        const int ib = idx_b[frame];
        indices_out[frame] = (float)(ia * CS + ib);

        const float* ca  = cb_a + (size_t)ia * CD;
        const float* cbv = cb_b + (size_t)ib * CD;
        const float* la  = latent + (size_t)b * 16 * TT + t;
#pragma unroll
        for (int c = 0; c < 8; ++c) {
            float qa = ca[c], qb = cbv[c];
            float za = la[(size_t)c * TT];
            float zb = la[(size_t)(8 + c) * TT];
            float da = za - qa, db = zb - qb;
            lsum += da * da + db * db;   // commitment == codebook loss (forward value)
            s_zq[c][tid]     = qa;
            s_zq[8 + c][tid] = qb;
        }
    }
#pragma unroll
    for (int s = 16; s >= 1; s >>= 1) lsum += __shfl_down(lsum, s, 32);
    if (tid < 64 && (tid & 31) == 0) {
        atomicAdd(&commit_loss[b], lsum * INV_MEAN);
        atomicAdd(&cbk_loss[b],    lsum * INV_MEAN);
    }
    __syncthreads();

    const int t     = tid & 63;
    const int o_sub = __builtin_amdgcn_readfirstlane(tid >> 6);  // wave-uniform 0..3
    float q[16];
#pragma unroll
    for (int c = 0; c < 16; ++c) q[c] = s_zq[c][t];

    float* za_out = zq_out + (size_t)b * DD * TT + t0 + t;
    float* zb_out = za_out + (size_t)512 * TT;
    for (int o = o_sub; o < 512; o += 4) {                 // o wave-uniform
        const float* wa = w_out_a + (size_t)o * CD;        // scalar loads
        float sa = b_out_a[o];
#pragma unroll
        for (int c = 0; c < 8; ++c) sa = fmaf(wa[c], q[c], sa);
        za_out[(size_t)o * TT] = sa;                       // coalesced over t

        const float* wb = w_out_b + (size_t)o * CD;
        float sb = b_out_b[o];
#pragma unroll
        for (int c = 0; c < 8; ++c) sb = fmaf(wb[c], q[8 + c], sb);
        zb_out[(size_t)o * TT] = sb;
    }
}

// ---------------------------------------------------------------------------
extern "C" void kernel_launch(void* const* d_in, const int* in_sizes, int n_in,
                              void* d_out, int out_size, void* d_ws, size_t ws_size,
                              hipStream_t stream) {
    const float* z       = (const float*)d_in[0];
    const float* w_in_a  = (const float*)d_in[1];
    const float* b_in_a  = (const float*)d_in[2];
    const float* w_in_b  = (const float*)d_in[3];
    const float* b_in_b  = (const float*)d_in[4];
    const float* w_out_a = (const float*)d_in[5];
    const float* b_out_a = (const float*)d_in[6];
    const float* w_out_b = (const float*)d_in[7];
    const float* b_out_b = (const float*)d_in[8];
    const float* cb_a    = (const float*)d_in[9];
    const float* cb_b    = (const float*)d_in[10];

    // d_out layout (flat f32, reference return order)
    float* out    = (float*)d_out;
    float* zq_out = out;                                   // B*D*T = 33,554,432
    float* commit = out + (size_t)BB * DD * TT;            // 8
    float* cbkl   = commit + BB;                           // 8
    float* idxf   = cbkl + BB;                             // B*T = 32,768
    float* latent = idxf + (size_t)BB * TT;                // B*16*T = 524,288

    // workspace layout (~1.5 MB)
    char* ws = (char*)d_ws;
    _Float16* encn  = (_Float16*)ws;                               // 1,048,576 B
    _Float16* cbn_a = (_Float16*)(ws + 1048576);                   //    65,536 B
    _Float16* cbn_b = (_Float16*)(ws + 1048576 + 65536);           //    65,536 B
    int*   idx_a    = (int*)(ws + 1048576 + 2 * 65536);            //   131,072 B
    int*   idx_b    = (int*)(ws + 1048576 + 2 * 65536 + 131072);   //   131,072 B
    float* wpack    = (float*)(ws + 1048576 + 2 * 65536 + 2 * 131072); // 65,536 B

    cbnorm_kernel<<<(2 * CS) / 256, 256, 0, stream>>>(cb_a, cb_b, cbn_a, cbn_b, commit);

    packw_kernel<<<(256 * 64) / 256, 256, 0, stream>>>(w_in_a, w_in_b, wpack);

    project_wmma_kernel<<<(NFRAMES / 16) / 8, 256, 0, stream>>>(
        z, wpack, b_in_a, b_in_b, latent, encn);

    vq_argmax_kernel<<<NFRAMES / 128, 256, 0, stream>>>(
        encn, cbn_a, cbn_b, idx_a, idx_b);

    output_kernel<<<dim3(TT / 64, BB), 256, 0, stream>>>(
        cb_a, cb_b, w_out_a, b_out_a, w_out_b, b_out_b,
        idx_a, idx_b, latent, zq_out, commit, cbkl, idxf);
}
